// BiLSTM_CRF_33552284516480
// MI455X (gfx1250) — matrix operand order repaired and checked by reading, not compiled, not run on hardware
//
#include <hip/hip_runtime.h>

// ---------------------------------------------------------------------------
// BiLSTM-CRF forward + Viterbi decode for gfx1250 (MI455X), wave32 + WMMA.
// B=32 T=512 E=512 HID=512 H=256 K=12, START=10 STOP=11.
// All dense math uses v_wmma_f32_16x16x32_f16 (f32 accumulate).
// A/B operands are pre-packed into the exact ISA 7.12.2 per-lane layouts so
// the GEMM/scan inner loops are pure b128 loads + WMMA.
// ---------------------------------------------------------------------------

typedef __attribute__((ext_vector_type(16))) _Float16 v16h;
typedef __attribute__((ext_vector_type(4)))  _Float16 v4h;
typedef __attribute__((ext_vector_type(8)))  float    v8f;

#define WMMA_F16(a, b, c) \
  __builtin_amdgcn_wmma_f32_16x16x32_f16(false, (a), false, (b), (short)0, (c), false, false)

static __device__ __forceinline__ float sigf(float x) { return 1.0f / (1.0f + __expf(-x)); }

static __device__ __forceinline__ v8f load8f(const float* __restrict__ p) {
  const float4* q = (const float4*)p;
  float4 x0 = q[0], x1 = q[1];
  v8f r;
  r[0] = x0.x; r[1] = x0.y; r[2] = x0.z; r[3] = x0.w;
  r[4] = x1.x; r[5] = x1.y; r[6] = x1.z; r[7] = x1.w;
  return r;
}

// Build the per-lane A operand (16 f16) for one 16x32 k-chunk from an f32 row.
// A layout (ISA 7.12.2): lane l, elem i -> (m=l%16, k=(i&7)+(i>=8?16:0)+(l>=16?8:0))
static __device__ __forceinline__ v16h load_a_from_f32row(const float* __restrict__ row,
                                                          int kbase, int lhi) {
  const float4* p0 = (const float4*)(row + kbase + (lhi << 3));
  const float4* p1 = (const float4*)(row + kbase + (lhi << 3) + 16);
  float4 x0 = p0[0], x1 = p0[1], y0 = p1[0], y1 = p1[1];
  v16h a;
  a[0] = (_Float16)x0.x;  a[1] = (_Float16)x0.y;  a[2] = (_Float16)x0.z;  a[3] = (_Float16)x0.w;
  a[4] = (_Float16)x1.x;  a[5] = (_Float16)x1.y;  a[6] = (_Float16)x1.z;  a[7] = (_Float16)x1.w;
  a[8] = (_Float16)y0.x;  a[9] = (_Float16)y0.y;  a[10] = (_Float16)y0.z; a[11] = (_Float16)y0.w;
  a[12] = (_Float16)y1.x; a[13] = (_Float16)y1.y; a[14] = (_Float16)y1.z; a[15] = (_Float16)y1.w;
  return a;
}

// ---------------------------------------------------------------------------
// Pack W [Nreal x Kdim] (row-major) into WMMA B-operand tile order:
//   dst[(((nt*(Kdim/32)+kc)*32)+lane)*16 + i] = f16(W[n*Kdim + k])
//   with n = nt*16 + lane%16, k = kc*32 + (i&7) + (i>=8 ? 16 : 0) + (lane>=16 ? 8 : 0).
// Rows n >= Nreal are zero-padded (used for W_out: 12 -> 16).
// ---------------------------------------------------------------------------
__global__ __launch_bounds__(256) void pack_b_f16(const float* __restrict__ W,
                                                  _Float16* __restrict__ dst,
                                                  int Ndim, int Nreal, int Kdim) {
  const int KC = Kdim >> 5;
  const int total = (Ndim >> 4) * KC * 512;
  for (int e = blockIdx.x * blockDim.x + threadIdx.x; e < total; e += gridDim.x * blockDim.x) {
    int i = e & 15;
    int lane = (e >> 4) & 31;
    int rest = e >> 9;
    int kc = rest % KC;
    int nt = rest / KC;
    int n = nt * 16 + (lane & 15);
    int k = kc * 32 + (i & 7) + ((i >= 8) ? 16 : 0) + ((lane >= 16) ? 8 : 0);
    float v = (n < Nreal) ? W[(long)n * Kdim + k] : 0.0f;
    dst[e] = (_Float16)v;
  }
}

// ---------------------------------------------------------------------------
// Phase 0b: gather embed[sentence] rows, convert to f16, and store directly in
// WMMA A-operand tile order: xpack[(((mt*16+kc)*32)+lane)*16 + i], m = b*T+t.
// Each thread produces 4 consecutive halves (one float4 read, one b64 write).
// ---------------------------------------------------------------------------
__global__ __launch_bounds__(256) void pack_x_f16(const int* __restrict__ sent,
                                                  const float* __restrict__ embed,
                                                  _Float16* __restrict__ xpack) {
  const int total4 = 1024 * 16 * 32 * 4;  // 2,097,152 groups of 4 halves
  for (int g = blockIdx.x * blockDim.x + threadIdx.x; g < total4; g += gridDim.x * blockDim.x) {
    int e = g << 2;
    int i0 = e & 15;
    int lane = (e >> 4) & 31;
    int rest = e >> 9;
    int kc = rest & 15;
    int mt = rest >> 4;
    int m = mt * 16 + (lane & 15);
    long row = (long)sent[m] * 512;
    int k = kc * 32 + ((lane >= 16) ? 8 : 0) + (i0 & 7) + ((i0 >= 8) ? 16 : 0);
    float4 v = *(const float4*)(embed + row + k);
    v4h d;
    d[0] = (_Float16)v.x; d[1] = (_Float16)v.y; d[2] = (_Float16)v.z; d[3] = (_Float16)v.w;
    *(v4h*)(xpack + e) = d;
  }
}

// ---------------------------------------------------------------------------
// Phase 1: xg2[t][g][b] = sum_e x[b*T+t, e] * w_ih[g, e] + bias[g]
// Output layout is transposed (batch innermost) so the sequential LSTM scan can
// seed its WMMA accumulators with contiguous float4 loads.
// grid = (1024 m-tiles, 2); 8 waves; each wave computes 4 N-tiles per A load.
// ---------------------------------------------------------------------------
__global__ __launch_bounds__(256) void gemm_xg_kernel(const _Float16* __restrict__ xpack,
                                                      const _Float16* __restrict__ pB,
                                                      const float* __restrict__ bias,
                                                      float* __restrict__ xg2) {
  const int lane = threadIdx.x & 31, wave = threadIdx.x >> 5;
  const int mt = blockIdx.x;
  const int ntg = (blockIdx.y * 8 + wave) * 4;  // first of 4 n-tiles
  const int lhi = lane >> 4, col = lane & 15;
  v8f acc[4] = {{}, {}, {}, {}};
#pragma unroll
  for (int kc = 0; kc < 16; ++kc) {
    v16h a = *(const v16h*)(xpack + (((long)mt * 16 + kc) * 32 + lane) * 16);
#pragma unroll
    for (int u = 0; u < 4; ++u) {
      v16h b = *(const v16h*)(pB + (((long)(ntg + u) * 16 + kc) * 32 + lane) * 16);
      acc[u] = WMMA_F16(a, b, acc[u]);
    }
  }
  const int bb = mt >> 5;                        // batch index (tiles never straddle b)
  const int tebase = (mt & 31) * 16 + lhi * 8;   // time index base for this lane half
#pragma unroll
  for (int u = 0; u < 4; ++u) {
    int g = (ntg + u) * 16 + col;
    float bv = bias[g];
#pragma unroll
    for (int r = 0; r < 8; ++r) {
      xg2[((long)(tebase + r) * 1024 + g) * 32 + bb] = acc[u][r] + bv;
    }
  }
}

// ---------------------------------------------------------------------------
// Phase 2: persistent bidirectional LSTM scan. grid.x = direction (0 fwd, 1 bwd),
// one 8-wave workgroup each. h kept in LDS pre-permuted into WMMA A layout
// ("hperm"), c kept in LDS f32. Per step: g = xg2[t] + h @ w_hh^T via WMMA,
// then gate nonlinearities, h written to feats[:, t, dir*256 + j].
// hperm flat index: [mt(2)][kc(8)][lane(32)][i(16)] f16.
// ---------------------------------------------------------------------------
__global__ __launch_bounds__(256) void lstm_scan_kernel(const float* __restrict__ xg_f,
                                                        const float* __restrict__ xg_b,
                                                        const _Float16* __restrict__ pWhh_f,
                                                        const _Float16* __restrict__ pWhh_b,
                                                        const float* __restrict__ h0,
                                                        const float* __restrict__ c0,
                                                        float* __restrict__ feats) {
  const int dir = blockIdx.x;
  const float* xg = dir ? xg_b : xg_f;
  const _Float16* pW = dir ? pWhh_b : pWhh_f;

  __shared__ _Float16 hperm[2 * 8 * 32 * 16];  // 16 KB
  __shared__ float c_lds[32 * 256];            // 32 KB

  const int tid = threadIdx.x;
  const int wave = tid >> 5, lane = tid & 31;
  const int lhi = lane >> 4, col = lane & 15;

  // Init c and hperm from h0/c0[dir].
  for (int e = tid; e < 32 * 256; e += 256) {
    int m = e >> 8, j = e & 255;
    c_lds[e] = c0[(dir * 32 + m) * 256 + j];
    float hv = h0[(dir * 32 + m) * 256 + j];
    int r = j & 31;
    int lh = (r >> 3) & 1;
    int ln = (m & 15) | (lh << 4);
    int ii = (r & 7) | ((r & 16) >> 1);
    hperm[(((m >> 4) * 8 + (j >> 5)) * 32 + ln) * 16 + ii] = (_Float16)hv;
  }
  __syncthreads();

  float hkeep[4][8];
  for (int t = 0; t < 512; ++t) {
    const int te = dir ? (511 - t) : t;
    const float* xgs = xg + (long)te * 32768;  // [1024 g][32 b] slab for this step

    // Prefetch next step's 128KB slab while this step's WMMAs run.
    if (t + 1 < 512) {
      const int tn = dir ? (te - 1) : (te + 1);
      const float* nx = xg + (long)tn * 32768;
#pragma unroll
      for (int q = 0; q < 4; ++q) {
        __builtin_prefetch(nx + (tid + q * 256) * 32, 0, 1);
      }
    }

#pragma unroll
    for (int cmb = 0; cmb < 4; ++cmb) {
      const int hb = (cmb & 1) * 8 + wave;  // hidden-column block 0..15
      const int mt = cmb >> 1;              // batch-row tile 0..1
      const int j = hb * 16 + col;          // hidden index of this lane's column
      const int mb0 = mt * 16 + lhi * 8;    // first of 8 consecutive batch rows
      // Seed accumulators with input pre-activations (contiguous along b).
      v8f ai = load8f(xgs + ((long)(0 * 256 + j)) * 32 + mb0);
      v8f af = load8f(xgs + ((long)(1 * 256 + j)) * 32 + mb0);
      v8f ag = load8f(xgs + ((long)(2 * 256 + j)) * 32 + mb0);
      v8f ao = load8f(xgs + ((long)(3 * 256 + j)) * 32 + mb0);
#pragma unroll
      for (int kc = 0; kc < 8; ++kc) {
        v16h a = *(const v16h*)(hperm + ((mt * 8 + kc) * 32 + lane) * 16);
        v16h bi = *(const v16h*)(pW + ((((0 * 16 + hb) * 8) + kc) * 32 + lane) * 16);
        v16h bf = *(const v16h*)(pW + ((((1 * 16 + hb) * 8) + kc) * 32 + lane) * 16);
        v16h bg = *(const v16h*)(pW + ((((2 * 16 + hb) * 8) + kc) * 32 + lane) * 16);
        v16h bo = *(const v16h*)(pW + ((((3 * 16 + hb) * 8) + kc) * 32 + lane) * 16);
        ai = WMMA_F16(a, bi, ai);
        af = WMMA_F16(a, bf, af);
        ag = WMMA_F16(a, bg, ag);
        ao = WMMA_F16(a, bo, ao);
      }
#pragma unroll
      for (int r = 0; r < 8; ++r) {  // gates + state update
        int mb = mb0 + r;
        float cm = c_lds[mb * 256 + j];
        float cn = sigf(af[r]) * cm + sigf(ai[r]) * tanhf(ag[r]);
        float hv = sigf(ao[r]) * tanhf(cn);
        c_lds[mb * 256 + j] = cn;
        feats[((long)mb * 512 + te) * 512 + dir * 256 + j] = hv;
        hkeep[cmb][r] = hv;
      }
    }
    __syncthreads();  // all waves done reading old hperm
#pragma unroll
    for (int cmb = 0; cmb < 4; ++cmb) {  // publish new h into hperm (A layout)
      const int hb = (cmb & 1) * 8 + wave;
      const int mt = cmb >> 1;
      const int j = hb * 16 + col;
      int r2 = j & 31;
      int lh = (r2 >> 3) & 1;
      int ii = (r2 & 7) | ((r2 & 16) >> 1);
#pragma unroll
      for (int r = 0; r < 8; ++r) {
        int mb = mt * 16 + r + lhi * 8;
        int ln = (mb & 15) | (lh << 4);
        hperm[(((mb >> 4) * 8 + (j >> 5)) * 32 + ln) * 16 + ii] = (_Float16)hkeep[cmb][r];
      }
    }
    __syncthreads();
  }
}

// ---------------------------------------------------------------------------
// Phase 3: emissions = feats @ W_out^T + b_out, N padded 12 -> 16.
// grid = 128 blocks * 8 waves = 1024 m-tiles.
// ---------------------------------------------------------------------------
__global__ __launch_bounds__(256) void proj_kernel(const float* __restrict__ feats,
                                                   const _Float16* __restrict__ pWout,
                                                   const float* __restrict__ b_out,
                                                   float* __restrict__ emis) {
  const int lane = threadIdx.x & 31, wave = threadIdx.x >> 5;
  const int mt = blockIdx.x * 8 + wave;
  const int lhi = lane >> 4, col = lane & 15;
  const float* arow = feats + (long)(mt * 16 + col) * 512;
  v8f acc = {};
#pragma unroll
  for (int kc = 0; kc < 16; ++kc) {
    v16h a = load_a_from_f32row(arow, kc * 32, lhi);
    v16h b = *(const v16h*)(pWout + ((long)kc * 32 + lane) * 16);
    acc = WMMA_F16(a, b, acc);
  }
  float bv = (col < 12) ? b_out[col] : 0.0f;
#pragma unroll
  for (int r = 0; r < 8; ++r) {
    int mg = mt * 16 + r + lhi * 8;
    emis[(long)mg * 16 + col] = acc[r] + bv;
  }
}

// ---------------------------------------------------------------------------
// Phase 4: Viterbi scan + backtrace. Single block, 384 threads (12 waves).
// scores[b,next,prev] = fv[b,prev] + trans[next,prev]; emit added after max.
// ---------------------------------------------------------------------------
__global__ __launch_bounds__(384) void viterbi_kernel(const float* __restrict__ emis,
                                                      const float* __restrict__ trans,
                                                      int* __restrict__ bptr,
                                                      float* __restrict__ out) {
  __shared__ float fv[2][32 * 12];
  __shared__ float tr[12 * 12];
  __shared__ float term[32 * 12];
  __shared__ int best[32];
  const int tid = threadIdx.x;
  if (tid < 144) tr[tid] = trans[tid];
  {
    int k = tid % 12;
    fv[0][tid] = (k == 10) ? 0.0f : -10000.0f;  // START=10
  }
  __syncthreads();

  for (int t = 0; t < 512; ++t) {
    const int cur = t & 1, nxt = cur ^ 1;
    const int b = tid / 12, nx = tid % 12;
    float bs = -3.4e38f;
    int bi = 0;
#pragma unroll
    for (int p = 0; p < 12; ++p) {
      float s = fv[cur][b * 12 + p] + tr[nx * 12 + p];
      if (s > bs) { bs = s; bi = p; }
    }
    bptr[((long)t * 32 + b) * 12 + nx] = bi;
    fv[nxt][tid] = bs + emis[((long)b * 512 + t) * 16 + nx];
    __syncthreads();
  }

  // terminal = fv + transitions[STOP=11, :]
  {
    int k = tid % 12;
    term[tid] = fv[0][tid] + tr[11 * 12 + k];
  }
  __syncthreads();
  if (tid < 32) {
    float bs = -3.4e38f;
    int bi = 0;
#pragma unroll
    for (int k = 0; k < 12; ++k) {
      float s = term[tid * 12 + k];
      if (s > bs) { bs = s; bi = k; }
    }
    best[tid] = bi;
  }
  __syncthreads();
  // path_score[i, j] = terminal[i, best[j]]  (faithful [B,B] output)
  for (int e = tid; e < 1024; e += 384) {
    int i = e >> 5, j = e & 31;
    out[e] = term[i * 12 + best[j]];
  }
  // backtrace per batch row
  if (tid < 32) {
    int b = tid;
    int tag = best[b];
    out[1024 + (long)b * 512 + 511] = (float)tag;
    for (int t = 511; t >= 1; --t) {
      tag = bptr[((long)t * 32 + b) * 12 + tag];
      out[1024 + (long)b * 512 + (t - 1)] = (float)tag;
    }
  }
}

// ---------------------------------------------------------------------------
// Host-side launcher.
// Inputs (setup_inputs order): 0 sentence(i32) 1 embed 2 w_ih_f 3 w_hh_f 4 b_f
// 5 w_ih_b 6 w_hh_b 7 b_b 8 h0 9 c0 10 W_out 11 b_out 12 transitions.
// ---------------------------------------------------------------------------
extern "C" void kernel_launch(void* const* d_in, const int* in_sizes, int n_in,
                              void* d_out, int out_size, void* d_ws, size_t ws_size,
                              hipStream_t stream) {
  (void)in_sizes; (void)n_in; (void)out_size; (void)ws_size;
  const int*   sent   = (const int*)d_in[0];
  const float* embed  = (const float*)d_in[1];
  const float* w_ih_f = (const float*)d_in[2];
  const float* w_hh_f = (const float*)d_in[3];
  const float* b_f    = (const float*)d_in[4];
  const float* w_ih_b = (const float*)d_in[5];
  const float* w_hh_b = (const float*)d_in[6];
  const float* b_b    = (const float*)d_in[7];
  const float* h0     = (const float*)d_in[8];
  const float* c0     = (const float*)d_in[9];
  const float* W_out  = (const float*)d_in[10];
  const float* b_out  = (const float*)d_in[11];
  const float* trans  = (const float*)d_in[12];
  float* out = (float*)d_out;

  char* ws = (char*)d_ws;
  const size_t MB = 1u << 20;
  _Float16* p_wih_f = (_Float16*)(ws + 0 * MB);            // 1 MB
  _Float16* p_wih_b = (_Float16*)(ws + 1 * MB);            // 1 MB
  _Float16* p_whh_f = (_Float16*)(ws + 2 * MB);            // 512 KB
  _Float16* p_whh_b = (_Float16*)(ws + 2 * MB + 512 * 1024);
  _Float16* p_wout  = (_Float16*)(ws + 3 * MB);            // 16 KB
  _Float16* xpack   = (_Float16*)(ws + 4 * MB);            // 16 MB
  float* xg_f  = (float*)(ws + 20 * MB);                   // 64 MB  [T][1024][32]
  float* xg_b  = (float*)(ws + 84 * MB);                   // 64 MB
  float* feats = (float*)(ws + 148 * MB);                  // 32 MB  [B][T][512]
  float* emis  = (float*)(ws + 180 * MB);                  // 1 MB   [B*T][16]
  int*   bptr  = (int*)(ws + 181 * MB);                    // 768 KB

  // Weight pre-swizzle into WMMA B-operand order.
  hipLaunchKernelGGL(pack_b_f16, dim3(128), dim3(256), 0, stream, w_ih_f, p_wih_f, 1024, 1024, 512);
  hipLaunchKernelGGL(pack_b_f16, dim3(128), dim3(256), 0, stream, w_ih_b, p_wih_b, 1024, 1024, 512);
  hipLaunchKernelGGL(pack_b_f16, dim3(128), dim3(256), 0, stream, w_hh_f, p_whh_f, 1024, 1024, 256);
  hipLaunchKernelGGL(pack_b_f16, dim3(128), dim3(256), 0, stream, w_hh_b, p_whh_b, 1024, 1024, 256);
  hipLaunchKernelGGL(pack_b_f16, dim3(16),  dim3(256), 0, stream, W_out,  p_wout,  16,   12,   512);

  // Embedding gather -> f16 A-operand tiles (shared by both directions).
  hipLaunchKernelGGL(pack_x_f16, dim3(2048), dim3(256), 0, stream, sent, embed, xpack);

  // Input pre-activations, stored transposed [t][gate][b] for the scan.
  hipLaunchKernelGGL(gemm_xg_kernel, dim3(1024, 2), dim3(256), 0, stream, xpack, p_wih_f, b_f, xg_f);
  hipLaunchKernelGGL(gemm_xg_kernel, dim3(1024, 2), dim3(256), 0, stream, xpack, p_wih_b, b_b, xg_b);

  // Persistent bidirectional recurrence (one workgroup per direction).
  hipLaunchKernelGGL(lstm_scan_kernel, dim3(2), dim3(256), 0, stream,
                     xg_f, xg_b, p_whh_f, p_whh_b, h0, c0, feats);

  // Emission projection (K=12 padded to 16).
  hipLaunchKernelGGL(proj_kernel, dim3(128), dim3(256), 0, stream, feats, p_wout, b_out, emis);

  // CRF Viterbi decode + backtrace.
  hipLaunchKernelGGL(viterbi_kernel, dim3(1), dim3(384), 0, stream, emis, trans, bptr, out);
}
